// SmolVLMWithExpertModel_31653908971641
// MI455X (gfx1250) — matrix operand order repaired and checked by reading, not compiled, not run on hardware
//
#include <hip/hip_runtime.h>
#include <hip/hip_bf16.h>

// ---------------- CDNA5 WMMA plumbing (gfx1250, wave32) ----------------
typedef __attribute__((ext_vector_type(16))) __bf16 v16bf;
typedef __attribute__((ext_vector_type(8)))  float  v8f;

union BFrag {
  v16bf v;
  unsigned int u[8];
  uint4 q[2];
  unsigned short s[16];
};

__device__ __forceinline__ unsigned short f2bf(float f) {
  unsigned int u = __float_as_uint(f);
  u += 0x7FFFu + ((u >> 16) & 1u);     // round-to-nearest-even
  return (unsigned short)(u >> 16);
}

__device__ __forceinline__ v8f wmma_bf16(v16bf a, v16bf b, v8f c) {
  // D = A(16x32 bf16) * B(32x16 bf16) + C(16x16 f32)
  return __builtin_amdgcn_wmma_f32_16x16x32_bf16(
      false, a, false, b, (short)0, c, false, false);
}

// A-fragment slot->K mapping used here (16-bit A 16x32, contiguous k matching the B loads):
//   lane half h = lane/16, pair v: k = 2v + 16h  (+chunk base)
#define A_K(v, half) (2 * (v) + 16 * (half))

#define LOG2E 1.44269504088896340736f
#define NEGF  (-1e30f)

// Problem constants
#define Dm   960
#define Hh   15
#define KVH_ 5
#define HD_  64
#define LV   2048
#define LE   64
#define LT   2112
#define NB   4

// ---------------- Kernel 0: weight fp32 (K x N) -> bf16 transposed (N x K) ----
__global__ void wtrans_kernel(const float* __restrict__ W,
                              unsigned short* __restrict__ WT, int K, int N) {
  int idx = blockIdx.x * 256 + threadIdx.x;
  if (idx >= K * N) return;
  int n = idx % N, k = idx / N;
  WT[(size_t)n * K + k] = f2bf(W[idx]);
}

// ---------------- Kernel 1: RMSNorm + QKV GEMM (WMMA bf16) + RoPE -------------
// Q out: [B][H][LT][HD], K out: [B][KVH][LT][HD], V out transposed: [B][KVH][HD][LT]
__global__ __launch_bounds__(256)
void qkv_rope_kernel(const float* __restrict__ hseg, const float* __restrict__ lnw,
                     const unsigned short* __restrict__ wqT,   // [960][960]
                     const unsigned short* __restrict__ wkT,   // [320][960]
                     const unsigned short* __restrict__ wvT,   // [320][960]
                     const int* __restrict__ pos,
                     unsigned short* __restrict__ Qo, unsigned short* __restrict__ Ko,
                     unsigned short* __restrict__ Vt, int Lseg, int seg_off) {
  __shared__ __align__(16) unsigned short xs[32][Dm];  // 60 KB normalized bf16 tile
  const int tiles = Lseg / 32;
  const int b = blockIdx.x / tiles;
  const int row0 = (blockIdx.x % tiles) * 32;
  const int t = threadIdx.x;

  // --- RMSNorm into LDS (8 threads per row) ---
  {
    const int row = t >> 3, sub = t & 7;
    const float* xr = hseg + ((size_t)b * Lseg + row0 + row) * Dm;
    float ss = 0.f;
    for (int d = sub * 120; d < sub * 120 + 120; ++d) { float v = xr[d]; ss += v * v; }
    ss += __shfl_xor(ss, 1, 8);
    ss += __shfl_xor(ss, 2, 8);
    ss += __shfl_xor(ss, 4, 8);
    float inv = rsqrtf(ss * (1.0f / (float)Dm) + 1e-6f);
    for (int d = sub * 120; d < sub * 120 + 120; ++d)
      xs[row][d] = f2bf(xr[d] * inv * lnw[d]);
  }
  __syncthreads();

  const int wave = t >> 5, lane = t & 31;
  const int nl = lane & 15, half = lane >> 4;

  // 50 units: 2 row-tiles x (15 Q-heads + 5 K-heads + 5 V-heads); 4 n-tiles each
  for (int u = wave; u < 50; u += 8) {
    const int rt = u / 25, sub = u % 25;
    int head, kind;
    if (sub < 15)      { head = sub;      kind = 0; }
    else if (sub < 20) { head = sub - 15; kind = 1; }
    else               { head = sub - 20; kind = 2; }
    const unsigned short* WT = (kind == 0) ? wqT : (kind == 1) ? wkT : wvT;

    v8f acc[4] = {};
    const int m = rt * 16 + nl;
#pragma unroll 2
    for (int kc = 0; kc < Dm / 32; ++kc) {
      BFrag a;
#pragma unroll
      for (int v = 0; v < 8; ++v)
        a.u[v] = *(const unsigned int*)&xs[m][kc * 32 + A_K(v, half)];
#pragma unroll
      for (int nt = 0; nt < 4; ++nt) {
        const unsigned short* bp =
            WT + (size_t)(head * 64 + nt * 16 + nl) * Dm + kc * 32 + 16 * half;
        BFrag bb; bb.q[0] = ((const uint4*)bp)[0]; bb.q[1] = ((const uint4*)bp)[1];
        acc[nt] = wmma_bf16(a.v, bb.v, acc[nt]);
      }
    }
    // RoPE for Q and K (pairs: col i with col i+32 within the 64-wide head)
    if (kind != 2) {
#pragma unroll
      for (int tt = 0; tt < 2; ++tt) {
        const float fi = (float)(tt * 16 + nl);
        const float inv_ts = exp2f(-fi * (13.287712379549449f / 32.0f)); // 10000^(-i/32)
#pragma unroll
        for (int r = 0; r < 8; ++r) {
          const int row = row0 + rt * 16 + r + 8 * half;
          float p = (float)pos[(size_t)b * Lseg + row];
          float sn, cs; __sincosf(p * inv_ts, &sn, &cs);
          float x1 = acc[tt][r], x2 = acc[tt + 2][r];
          acc[tt][r]     = x1 * cs - x2 * sn;
          acc[tt + 2][r] = x2 * cs + x1 * sn;
        }
      }
    }
    const int tok0 = seg_off + row0 + rt * 16;
    if (kind == 0) {       // Q, pre-scaled by HD^-0.5
      unsigned short* qp = Qo + (((size_t)b * Hh + head) * LT) * HD_;
#pragma unroll
      for (int tt = 0; tt < 4; ++tt)
#pragma unroll
        for (int r = 0; r < 8; ++r)
          qp[(size_t)(tok0 + r + 8 * half) * HD_ + tt * 16 + nl] = f2bf(acc[tt][r] * 0.125f);
    } else if (kind == 1) { // K
      unsigned short* kp = Ko + (((size_t)b * KVH_ + head) * LT) * HD_;
#pragma unroll
      for (int tt = 0; tt < 4; ++tt)
#pragma unroll
        for (int r = 0; r < 8; ++r)
          kp[(size_t)(tok0 + r + 8 * half) * HD_ + tt * 16 + nl] = f2bf(acc[tt][r]);
    } else {                // V stored transposed [HD][LT]
      unsigned short* vp = Vt + (((size_t)b * KVH_ + head) * HD_) * LT;
#pragma unroll
      for (int tt = 0; tt < 4; ++tt)
#pragma unroll
        for (int r = 0; r < 8; ++r)
          vp[(size_t)(tt * 16 + nl) * LT + tok0 + r + 8 * half] = f2bf(acc[tt][r]);
    }
  }
}

// ---------------- Kernel 2: flash attention, 1 wave per (b, h, 16-row q tile) --
__global__ __launch_bounds__(32)
void flash_kernel(const unsigned short* __restrict__ Q,
                  const unsigned short* __restrict__ K,
                  const unsigned short* __restrict__ Vt,
                  unsigned short* __restrict__ attn) {   // [B][LT][H*HD] bf16
  __shared__ __align__(16) unsigned short Pl[16][32];    // P re-swizzle bounce
  const int qt = blockIdx.x % (LT / 16);
  const int h  = (blockIdx.x / (LT / 16)) % Hh;
  const int b  = blockIdx.x / ((LT / 16) * Hh);
  const int kvh = h / (Hh / KVH_);
  const int lane = threadIdx.x, nl = lane & 15, half = lane >> 4;
  const int q0 = qt * 16;

  const unsigned short* qp = Q + (((size_t)b * Hh + h) * LT + q0) * HD_;
  BFrag qa[2];
#pragma unroll
  for (int c = 0; c < 2; ++c)
#pragma unroll
    for (int v = 0; v < 8; ++v)
      qa[c].u[v] = *(const unsigned int*)(qp + (size_t)nl * HD_ + c * 32 + A_K(v, half));

  const unsigned short* kp = K  + ((size_t)b * KVH_ + kvh) * LT * HD_;
  const unsigned short* vp = Vt + ((size_t)b * KVH_ + kvh) * HD_ * LT;

  float mi[8], li[8];
#pragma unroll
  for (int r = 0; r < 8; ++r) { mi[r] = NEGF; li[r] = 0.f; }
  v8f acc[4] = {};

  const bool is_exp = (q0 >= LV);
  const int kv_end = is_exp ? (q0 + 16) : LV;

  for (int j = 0; j < kv_end; j += 32) {
    if (j + 32 < kv_end)
      __builtin_prefetch(kp + (size_t)(j + 32 + nl) * HD_, 0, 3);  // global_prefetch

    v8f s0 = {}, s1 = {};
#pragma unroll
    for (int c = 0; c < 2; ++c) {
      const unsigned short* b0 = kp + (size_t)(j + nl) * HD_ + c * 32 + 16 * half;
      BFrag k0; k0.q[0] = ((const uint4*)b0)[0]; k0.q[1] = ((const uint4*)b0)[1];
      s0 = wmma_bf16(qa[c].v, k0.v, s0);
      const unsigned short* b1 = kp + (size_t)(j + 16 + nl) * HD_ + c * 32 + 16 * half;
      BFrag k1; k1.q[0] = ((const uint4*)b1)[0]; k1.q[1] = ((const uint4*)b1)[1];
      s1 = wmma_bf16(qa[c].v, k1.v, s1);
    }
    if (is_exp && (j + 31 > q0)) {            // causal mask inside expert block
#pragma unroll
      for (int r = 0; r < 8; ++r) {
        const int qrow = q0 + r + 8 * half;
        if (j + nl > qrow)      s0[r] = NEGF;
        if (j + 16 + nl > qrow) s1[r] = NEGF;
      }
    }
    // online softmax (row stats replicated across the 16 lanes of each half)
    float p0[8], p1[8], corr[8];
#pragma unroll
    for (int r = 0; r < 8; ++r) {
      float mx = fmaxf(s0[r], s1[r]);
      mx = fmaxf(mx, __shfl_xor(mx, 1, 16));
      mx = fmaxf(mx, __shfl_xor(mx, 2, 16));
      mx = fmaxf(mx, __shfl_xor(mx, 4, 16));
      mx = fmaxf(mx, __shfl_xor(mx, 8, 16));
      float mnew = fmaxf(mi[r], mx);
      corr[r] = exp2f((mi[r] - mnew) * LOG2E);
      p0[r] = exp2f((s0[r] - mnew) * LOG2E);
      p1[r] = exp2f((s1[r] - mnew) * LOG2E);
      float rs = p0[r] + p1[r];
      rs += __shfl_xor(rs, 1, 16);
      rs += __shfl_xor(rs, 2, 16);
      rs += __shfl_xor(rs, 4, 16);
      rs += __shfl_xor(rs, 8, 16);
      li[r] = li[r] * corr[r] + rs;
      mi[r] = mnew;
    }
#pragma unroll
    for (int tt = 0; tt < 4; ++tt)
#pragma unroll
      for (int r = 0; r < 8; ++r) acc[tt][r] = acc[tt][r] * corr[r];

    // C-layout P -> LDS -> A-fragment layout
#pragma unroll
    for (int r = 0; r < 8; ++r) {
      Pl[r + 8 * half][nl]      = f2bf(p0[r]);
      Pl[r + 8 * half][16 + nl] = f2bf(p1[r]);
    }
    __syncthreads();
    BFrag pa;
#pragma unroll
    for (int v = 0; v < 8; ++v)
      pa.u[v] = *(const unsigned int*)&Pl[nl][A_K(v, half)];
    __syncthreads();

#pragma unroll
    for (int tt = 0; tt < 4; ++tt) {
      const unsigned short* bp = vp + (size_t)(tt * 16 + nl) * LT + j + 16 * half;
      BFrag vb; vb.q[0] = ((const uint4*)bp)[0]; vb.q[1] = ((const uint4*)bp)[1];
      acc[tt] = wmma_bf16(pa.v, vb.v, acc[tt]);
    }
  }
  unsigned short* ap = attn + ((size_t)b * LT + q0) * (Hh * HD_) + h * HD_;
#pragma unroll
  for (int r = 0; r < 8; ++r) {
    float inv = 1.0f / li[r];
#pragma unroll
    for (int tt = 0; tt < 4; ++tt)
      ap[(size_t)(r + 8 * half) * (Hh * HD_) + tt * 16 + nl] = f2bf(acc[tt][r] * inv);
  }
}

// ---------------- Kernel 3: out-projection (WMMA) + residual, fp32 out --------
__global__ __launch_bounds__(256)
void proj_kernel(const unsigned short* __restrict__ attn,
                 const unsigned short* __restrict__ woT_v,
                 const unsigned short* __restrict__ woT_e,
                 const float* __restrict__ hv, const float* __restrict__ he,
                 float* __restrict__ out) {
  __shared__ __align__(16) unsigned short as[32][Dm];  // 60 KB attn tile
  const int g = blockIdx.x;
  const int b = (g * 32) / LT;
  const int tok0 = (g * 32) % LT;
  const bool is_exp = tok0 >= LV;
  const unsigned short* WT = is_exp ? woT_e : woT_v;
  const float* res = is_exp ? (he + ((size_t)b * LE + (tok0 - LV)) * Dm)
                            : (hv + ((size_t)b * LV + tok0) * Dm);
  const unsigned short* ap = attn + ((size_t)b * LT + tok0) * Dm;
  {
    const uint4* src = (const uint4*)ap;
    uint4* dst = (uint4*)&as[0][0];
    for (int i = threadIdx.x; i < 32 * Dm / 8; i += 256) dst[i] = src[i];
  }
  __syncthreads();
  const int wave = threadIdx.x >> 5, lane = threadIdx.x & 31;
  const int nl = lane & 15, half = lane >> 4;
  float* outp = out + ((size_t)b * LT + tok0) * Dm;

  for (int u = wave; u < 120; u += 8) {   // 2 row-tiles x 60 n-tiles
    const int rt = u / 60, nt = u % 60;
    v8f c = {};
#pragma unroll 2
    for (int kc = 0; kc < Dm / 32; ++kc) {
      BFrag a;
#pragma unroll
      for (int v = 0; v < 8; ++v)
        a.u[v] = *(const unsigned int*)&as[rt * 16 + nl][kc * 32 + A_K(v, half)];
      const unsigned short* bp = WT + (size_t)(nt * 16 + nl) * Dm + kc * 32 + 16 * half;
      BFrag bb; bb.q[0] = ((const uint4*)bp)[0]; bb.q[1] = ((const uint4*)bp)[1];
      c = wmma_bf16(a.v, bb.v, c);
    }
#pragma unroll
    for (int r = 0; r < 8; ++r) {
      const int row = rt * 16 + r + 8 * half;
      const int d = nt * 16 + nl;
      outp[(size_t)row * Dm + d] = res[(size_t)row * Dm + d] + c[r];
    }
  }
}

// ---------------- host launch -------------------------------------------------
extern "C" void kernel_launch(void* const* d_in, const int* in_sizes, int n_in,
                              void* d_out, int out_size, void* d_ws, size_t ws_size,
                              hipStream_t stream) {
  const float* hv   = (const float*)d_in[0];
  const float* he   = (const float*)d_in[1];
  const float* ln_v = (const float*)d_in[2];
  const float* wq_v = (const float*)d_in[3];
  const float* wk_v = (const float*)d_in[4];
  const float* wv_v = (const float*)d_in[5];
  const float* wo_v = (const float*)d_in[6];
  const float* ln_e = (const float*)d_in[7];
  const float* wq_e = (const float*)d_in[8];
  const float* wk_e = (const float*)d_in[9];
  const float* wv_e = (const float*)d_in[10];
  const float* wo_e = (const float*)d_in[11];
  const int* pos_v  = (const int*)d_in[12];
  const int* pos_e  = (const int*)d_in[13];
  float* out = (float*)d_out;

  char* ws = (char*)d_ws;
  size_t off = 0;
  auto take = [&](size_t elems) {
    unsigned short* p = (unsigned short*)(ws + off);
    off += (elems * sizeof(unsigned short) + 255) & ~(size_t)255;
    return p;
  };
  unsigned short* wqT_v = take(960 * 960);
  unsigned short* wkT_v = take(320 * 960);
  unsigned short* wvT_v = take(320 * 960);
  unsigned short* woT_v = take(960 * 960);
  unsigned short* wqT_e = take(960 * 960);
  unsigned short* wkT_e = take(320 * 960);
  unsigned short* wvT_e = take(320 * 960);
  unsigned short* woT_e = take(960 * 960);
  unsigned short* Qb   = take((size_t)NB * Hh * LT * HD_);
  unsigned short* Kb   = take((size_t)NB * KVH_ * LT * HD_);
  unsigned short* Vtb  = take((size_t)NB * KVH_ * HD_ * LT);
  unsigned short* attn = take((size_t)NB * LT * Dm);

  auto wt = [&](const float* w, unsigned short* wT, int K, int N) {
    int n = K * N;
    wtrans_kernel<<<(n + 255) / 256, 256, 0, stream>>>(w, wT, K, N);
  };
  wt(wq_v, wqT_v, 960, 960); wt(wk_v, wkT_v, 960, 320);
  wt(wv_v, wvT_v, 960, 320); wt(wo_v, woT_v, 960, 960);
  wt(wq_e, wqT_e, 960, 960); wt(wk_e, wkT_e, 960, 320);
  wt(wv_e, wvT_e, 960, 320); wt(wo_e, woT_e, 960, 960);

  qkv_rope_kernel<<<NB * (LV / 32), 256, 0, stream>>>(
      hv, ln_v, wqT_v, wkT_v, wvT_v, pos_v, Qb, Kb, Vtb, LV, 0);
  qkv_rope_kernel<<<NB * (LE / 32), 256, 0, stream>>>(
      he, ln_e, wqT_e, wkT_e, wvT_e, pos_e, Qb, Kb, Vtb, LE, LV);

  flash_kernel<<<NB * Hh * (LT / 16), 32, 0, stream>>>(Qb, Kb, Vtb, attn);

  proj_kernel<<<NB * (LT / 32), 256, 0, stream>>>(attn, woT_v, woT_e, hv, he, out);
}
